// DGCNN_PyG_9448928051625
// MI455X (gfx1250) — matrix operand order repaired and checked by reading, not compiled, not run on hardware
//
#include <hip/hip_runtime.h>

// ---------------------------------------------------------------------------
// DGCNN forward for MI455X (gfx1250, wave32, WMMA).
// GEMMs: f16 x f16 -> f32 via v_wmma_f32_16x16x32_f16, with double-buffered
// LDS tiles staged by CDNA5 async-to-LDS copies (global_load_async_to_lds_b128,
// tracked by ASYNCcnt) so global fetch overlaps WMMA compute.
// ---------------------------------------------------------------------------

typedef __attribute__((ext_vector_type(16))) _Float16 v16h;
typedef __attribute__((ext_vector_type(8)))  _Float16 half8;
typedef __attribute__((ext_vector_type(8)))  float    v8f;

#define BC 8
#define NP 1024
#define KN 20
#define NSPLIT 64

static constexpr long long ME = 1LL * BC * NP * KN;   // 163840 edges
static constexpr long long MP = 1LL * BC * NP;        // 8192 points

// CDNA5 async copy: 16 bytes global -> LDS, no VGPR round-trip (ASYNCcnt).
__device__ __forceinline__ void async_copy_b128(const _Float16* lds_dst,
                                                const _Float16* gsrc) {
  const unsigned lds_off = (unsigned)(unsigned long long)(uintptr_t)lds_dst;
  asm volatile("global_load_async_to_lds_b128 %0, %1, off"
               :: "v"(lds_off), "v"((unsigned long long)(uintptr_t)gsrc)
               : "memory");
}
__device__ __forceinline__ void wait_async0() {
  asm volatile("s_wait_asynccnt 0" ::: "memory");
}

// ---------------------------------------------------------------------------
// KNN: per point, 20 nearest in-cloud neighbors by squared distance.
// One block = 256 points of one cloud; whole cloud staged in LDS (12 KB).
// ---------------------------------------------------------------------------
__global__ __launch_bounds__(256) void knn_kernel(const float* __restrict__ pos,
                                                  int* __restrict__ knn) {
  __shared__ float sx[NP * 3];
  const int b = blockIdx.y;
  const int i = blockIdx.x * 256 + threadIdx.x;
  for (int t = threadIdx.x; t < NP * 3; t += 256) sx[t] = pos[b * NP * 3 + t];
  __syncthreads();
  const float x0 = sx[i * 3 + 0], x1 = sx[i * 3 + 1], x2 = sx[i * 3 + 2];
  float bd[KN];
  int   bi[KN];
#pragma unroll
  for (int k = 0; k < KN; ++k) { bd[k] = 3.4e38f; bi[k] = 0; }
  for (int j = 0; j < NP; ++j) {
    if (j == i) continue;
    const float d0 = sx[j * 3 + 0] - x0;
    const float d1 = sx[j * 3 + 1] - x1;
    const float d2 = sx[j * 3 + 2] - x2;
    const float d  = d0 * d0 + d1 * d1 + d2 * d2;
    if (d < bd[KN - 1]) {
      int p = KN - 1;
      while (p > 0 && bd[p - 1] > d) { bd[p] = bd[p - 1]; bi[p] = bi[p - 1]; --p; }
      bd[p] = d; bi[p] = j;
    }
  }
  int* o = knn + (long long)(b * NP + i) * KN;
#pragma unroll
  for (int k = 0; k < KN; ++k) o[k] = bi[k];
}

// ---------------------------------------------------------------------------
// Small conversion kernels
// ---------------------------------------------------------------------------
__global__ __launch_bounds__(256) void cvt_f32_to_f16(const float* __restrict__ src,
                                                      _Float16* __restrict__ dst,
                                                      long long n) {
  long long t = (long long)blockIdx.x * 256 + threadIdx.x;
  if (t < n) dst[t] = (_Float16)src[t];
}

// W [Kd, Nd] f32 row-major -> Wt [Nd, Kpad] f16 row-major (zero-padded K).
__global__ __launch_bounds__(256) void cvt_weight_t(const float* __restrict__ W,
                                                    _Float16* __restrict__ Wt,
                                                    int Kd, int Nd, int Kpad) {
  long long t = (long long)blockIdx.x * 256 + threadIdx.x;
  if (t >= (long long)Nd * Kpad) return;
  const int k = (int)(t % Kpad);
  const int n = (int)(t / Kpad);
  Wt[t] = (k < Kd) ? (_Float16)W[(size_t)k * Nd + n] : (_Float16)0.0f;
}

// ---------------------------------------------------------------------------
// Edge feature build: Ae[e, 0:Cin] = x_i ; Ae[e, Cin:2Cin] = x_j - x_i ; pad 0.
// ---------------------------------------------------------------------------
__global__ __launch_bounds__(256) void build_edges(const _Float16* __restrict__ x,
                                                   const int* __restrict__ knn,
                                                   _Float16* __restrict__ Ae,
                                                   int Cin, int Cpad) {
  long long t = (long long)blockIdx.x * 256 + threadIdx.x;
  if (t >= ME * (long long)Cpad) return;
  const int       c  = (int)(t % Cpad);
  const long long e  = t / Cpad;
  const long long pi = e / KN;
  const int       b  = (int)(pi >> 10);
  _Float16 v;
  if (c < Cin) {
    v = x[(size_t)pi * Cin + c];
  } else if (c < 2 * Cin) {
    const int       cc = c - Cin;
    const long long pj = ((long long)b << 10) + knn[e];
    v = (_Float16)((float)x[(size_t)pj * Cin + cc] - (float)x[(size_t)pi * Cin + cc]);
  } else {
    v = (_Float16)0.0f;
  }
  Ae[t] = v;
}

// ---------------------------------------------------------------------------
// WMMA GEMM: C[M,Nd] = A[M,Kd](f16) @ W[Kd,Nd] + bias, W given transposed as
// Wt[Nd,Kd] f16. Tile 128x64x32, 8 waves (4x2), 2x2 16x16 subtiles per wave.
// Double-buffered LDS, async-to-LDS staging, one barrier per K-step.
// Requires M%128==0, Nd%64==0, Kd%32==0.
// ---------------------------------------------------------------------------
__global__ __launch_bounds__(256) void gemm_f16_wmma(const _Float16* __restrict__ A,
                                                     const _Float16* __restrict__ Wt,
                                                     const float* __restrict__ bias,
                                                     float* __restrict__ C,
                                                     int M, int Nd, int Kd) {
  __shared__ _Float16 sA[2][128 * 32];
  __shared__ _Float16 sB[2][64 * 32];
  const int tid  = threadIdx.x;
  const int lane = tid & 31;
  const int wave = tid >> 5;
  const int wm   = wave >> 1;   // 0..3 (row group)
  const int wn   = wave & 1;    // 0..1 (col group)
  const int rl   = lane & 15;
  const int hs   = lane >> 4;   // half select
  const int rowBase = blockIdx.y * 128;
  const int colBase = blockIdx.x * 64;

  const int ka = hs ? 8 : 0;    // A: per-lane K chunk base
  const int kb = hs ? 8 : 0;    // B: per-lane K chunk base

  // Per-thread staging assignment (16B chunks): A tile 128x32 -> 512 chunks
  // (2/thread), B tile 64x32 -> 256 chunks (1/thread).
  const int chA  = tid * 2;
  const int rA0  = chA >> 2,       cA0 = (chA & 3) * 8;
  const int rA1  = (chA + 1) >> 2, cA1 = ((chA + 1) & 3) * 8;
  const int rB   = tid >> 2,       cB  = (tid & 3) * 8;

  auto stage = [&](int buf, int k0) {
    async_copy_b128(&sA[buf][rA0 * 32 + cA0], &A[(size_t)(rowBase + rA0) * Kd + k0 + cA0]);
    async_copy_b128(&sA[buf][rA1 * 32 + cA1], &A[(size_t)(rowBase + rA1) * Kd + k0 + cA1]);
    async_copy_b128(&sB[buf][rB  * 32 + cB ], &Wt[(size_t)(colBase + rB) * Kd + k0 + cB]);
  };

  stage(0, 0);
  wait_async0();
  __syncthreads();

  v8f acc[2][2] = {};
  int buf = 0;
  for (int k0 = 0; k0 < Kd; k0 += 32) {
    // Prefetch next K-tile into the other buffer while we compute this one.
    if (k0 + 32 < Kd) stage(buf ^ 1, k0 + 32);

    v16h af[2], bf[2];
#pragma unroll
    for (int sm = 0; sm < 2; ++sm) {
      // A fragment: row = wm*32 + sm*16 + rl ; K halves [ka..ka+7],[ka+16..ka+23]
      const _Float16* p = &sA[buf][(wm * 32 + sm * 16 + rl) * 32 + ka];
      union { v16h v; half8 h[2]; } u;
      u.h[0] = *(const half8*)(p);
      u.h[1] = *(const half8*)(p + 16);
      af[sm] = u.v;
    }
#pragma unroll
    for (int sn = 0; sn < 2; ++sn) {
      // B fragment: col n = wn*32 + sn*16 + rl ; K halves [kb..kb+7],[kb+16..kb+23]
      const _Float16* p = &sB[buf][(wn * 32 + sn * 16 + rl) * 32 + kb];
      union { v16h v; half8 h[2]; } u;
      u.h[0] = *(const half8*)(p);
      u.h[1] = *(const half8*)(p + 16);
      bf[sn] = u.v;
    }
#pragma unroll
    for (int sm = 0; sm < 2; ++sm)
#pragma unroll
      for (int sn = 0; sn < 2; ++sn)
        acc[sm][sn] = __builtin_amdgcn_wmma_f32_16x16x32_f16(
            false, af[sm], false, bf[sn], (short)0, acc[sm][sn], false, false);

    // Next buffer fully landed in LDS + all waves done reading current buffer.
    wait_async0();
    __syncthreads();
    buf ^= 1;
  }

  // Epilogue: C layout — VGPR q holds row q (lanes 0-15) / q+8 (lanes 16-31).
#pragma unroll
  for (int sm = 0; sm < 2; ++sm) {
#pragma unroll
    for (int sn = 0; sn < 2; ++sn) {
      const int   gc  = colBase + wn * 32 + sn * 16 + rl;
      const float bv  = bias ? bias[gc] : 0.0f;
      const int   gr0 = rowBase + wm * 32 + sm * 16 + (hs ? 8 : 0);
#pragma unroll
      for (int q = 0; q < 8; ++q)
        C[(size_t)(gr0 + q) * Nd + gc] = acc[sm][sn][q] + bv;
    }
  }
}

// ---------------------------------------------------------------------------
// BatchNorm stats over M rows (biased var), deterministic split-partials.
// ---------------------------------------------------------------------------
__global__ __launch_bounds__(256) void bn_stat_partial(const float* __restrict__ H,
                                                       float* __restrict__ psum,
                                                       float* __restrict__ pss,
                                                       int M, int Cch) {
  const int c = blockIdx.x * 256 + threadIdx.x;
  if (c >= Cch) return;
  const int rows = (M + NSPLIT - 1) / NSPLIT;
  const int r0   = blockIdx.y * rows;
  const int r1   = min(M, r0 + rows);
  float s = 0.f, ss = 0.f;
  for (int r = r0; r < r1; ++r) {
    const float v = H[(size_t)r * Cch + c];
    s += v; ss += v * v;
  }
  psum[(size_t)blockIdx.y * Cch + c] = s;
  pss [(size_t)blockIdx.y * Cch + c] = ss;
}

__global__ __launch_bounds__(256) void bn_finalize(const float* __restrict__ psum,
                                                   const float* __restrict__ pss,
                                                   float* __restrict__ mean,
                                                   float* __restrict__ istd,
                                                   int M, int Cch) {
  const int c = blockIdx.x * 256 + threadIdx.x;
  if (c >= Cch) return;
  float s = 0.f, ss = 0.f;
  for (int i = 0; i < NSPLIT; ++i) {
    s  += psum[(size_t)i * Cch + c];
    ss += pss [(size_t)i * Cch + c];
  }
  const float m = s / (float)M;
  const float v = ss / (float)M - m * m;
  mean[c] = m;
  istd[c] = rsqrtf(fmaxf(v, 0.f) + 1e-5f);
}

__global__ __launch_bounds__(256) void bn_apply_lrelu_f16(const float* __restrict__ H,
                                                          const float* __restrict__ mean,
                                                          const float* __restrict__ istd,
                                                          const float* __restrict__ g,
                                                          const float* __restrict__ be,
                                                          _Float16* __restrict__ out,
                                                          long long total, int Cch) {
  long long t = (long long)blockIdx.x * 256 + threadIdx.x;
  if (t >= total) return;
  const int c = (int)(t % Cch);
  float y = (H[t] - mean[c]) * istd[c] * g[c] + be[c];
  y = y > 0.f ? y : 0.2f * y;
  out[t] = (_Float16)y;
}

// ---------------------------------------------------------------------------
// Max over K neighbors; also scatter into the concat buffer xc (stride 512).
// ---------------------------------------------------------------------------
__global__ __launch_bounds__(256) void max_over_k(const _Float16* __restrict__ E,
                                                  _Float16* __restrict__ xout,
                                                  _Float16* __restrict__ xc,
                                                  int Cch, int xcOff) {
  long long t = (long long)blockIdx.x * 256 + threadIdx.x;
  if (t >= MP * (long long)Cch) return;
  const int       c = (int)(t % Cch);
  const long long p = t / Cch;
  const _Float16* base = E + (size_t)p * KN * Cch + c;
  float m = -3.4e38f;
#pragma unroll 4
  for (int k = 0; k < KN; ++k) m = fmaxf(m, (float)base[(size_t)k * Cch]);
  const _Float16 hv = (_Float16)m;
  xout[(size_t)p * Cch + c]       = hv;
  xc[(size_t)p * 512 + xcOff + c] = hv;
}

// ---------------------------------------------------------------------------
// Segment max + mean pooling over each cloud: h [8192,1024] -> pooled [8,2048].
// ---------------------------------------------------------------------------
__global__ __launch_bounds__(256) void pool_max_mean(const _Float16* __restrict__ h,
                                                     float* __restrict__ pooled) {
  const int b = blockIdx.y;
  const int c = blockIdx.x * 256 + threadIdx.x;   // < 1024
  float mx = -3.4e38f, sm = 0.f;
  for (int r = 0; r < NP; ++r) {
    const float v = (float)h[(size_t)(b * NP + r) * 1024 + c];
    mx = fmaxf(mx, v); sm += v;
  }
  pooled[b * 2048 + c]        = mx;
  pooled[b * 2048 + 1024 + c] = sm * (1.0f / NP);
}

// ---------------------------------------------------------------------------
// Tiny tail layers (M = 8 rows): scalar GEMM + BN-over-8-rows + LeakyReLU.
// ---------------------------------------------------------------------------
__global__ __launch_bounds__(256) void lin_small(const float* __restrict__ X,
                                                 const float* __restrict__ W,
                                                 const float* __restrict__ bias,
                                                 float* __restrict__ Z,
                                                 int Kd, int Nd) {
  const int t = blockIdx.x * 256 + threadIdx.x;
  if (t >= 8 * Nd) return;
  const int n = t % Nd, r = t / Nd;
  float a = bias[n];
  for (int k = 0; k < Kd; ++k) a += X[r * Kd + k] * W[(size_t)k * Nd + n];
  Z[t] = a;
}

__global__ __launch_bounds__(256) void bn_tail_apply(const float* __restrict__ Z,
                                                     const float* __restrict__ g,
                                                     const float* __restrict__ be,
                                                     float* __restrict__ Y, int Nd) {
  const int c = blockIdx.x * 256 + threadIdx.x;
  if (c >= Nd) return;
  float s = 0.f, ss = 0.f;
  for (int r = 0; r < 8; ++r) { const float v = Z[r * Nd + c]; s += v; ss += v * v; }
  const float m  = s * 0.125f;
  const float va = ss * 0.125f - m * m;
  const float is = rsqrtf(fmaxf(va, 0.f) + 1e-5f);
  for (int r = 0; r < 8; ++r) {
    const float y = (Z[r * Nd + c] - m) * is * g[c] + be[c];
    Y[r * Nd + c] = y > 0.f ? y : 0.2f * y;
  }
}

// ---------------------------------------------------------------------------
// Host orchestration
// ---------------------------------------------------------------------------
extern "C" void kernel_launch(void* const* d_in, const int* in_sizes, int n_in,
                              void* d_out, int out_size, void* d_ws, size_t ws_size,
                              hipStream_t stream) {
  (void)in_sizes; (void)n_in; (void)out_size; (void)ws_size;
  const float* pos = (const float*)d_in[0];
  // d_in layout: 0 pos, 1 batch, 2..33 conv1..conv4 (W1,b1,g1,be1,W2,b2,g2,be2),
  // 34/35 lin1.W/b, 36/37 bn1.g/b, 38/39 lin2, 40/41 bn2, 42/43 lin3, 44/45 bn3, 46/47 lin4

  char*  ws  = (char*)d_ws;
  size_t off = 0;
  auto alloc = [&](size_t bytes) -> char* {
    char* p = ws + off;
    off += (bytes + 255) & ~(size_t)255;
    return p;
  };
  auto cdiv = [](long long a, long long b) -> unsigned { return (unsigned)((a + b - 1) / b); };

  int*       idx   = (int*)alloc(ME * 4);
  _Float16*  x0    = (_Float16*)alloc((size_t)MP * 3 * 2);
  _Float16*  xbuf  = (_Float16*)alloc((size_t)MP * 256 * 2);
  _Float16*  xc    = (_Float16*)alloc((size_t)MP * 512 * 2);

  const int cin[4]  = {3, 64, 64, 128};     // input channels per edge-conv
  const int kp1[4]  = {32, 128, 128, 256};  // padded 2*Cin (conv1: 6 -> 32)
  const int cout[4] = {64, 64, 128, 256};

  _Float16* w1t[4];
  _Float16* w2t[4];
  for (int l = 0; l < 4; ++l) {
    w1t[l] = (_Float16*)alloc((size_t)cout[l] * kp1[l] * 2);
    w2t[l] = (_Float16*)alloc((size_t)cout[l] * cout[l] * 2);
  }
  _Float16* wl1t   = (_Float16*)alloc((size_t)1024 * 512 * 2);

  float* psum   = (float*)alloc((size_t)NSPLIT * 1024 * 4);
  float* pss    = (float*)alloc((size_t)NSPLIT * 1024 * 4);
  float* meanb  = (float*)alloc(1024 * 4);
  float* istdb  = (float*)alloc(1024 * 4);
  float* pooled = (float*)alloc(8 * 2048 * 4);
  float* z2     = (float*)alloc(8 * 512 * 4);
  float* a2     = (float*)alloc(8 * 512 * 4);
  float* z3     = (float*)alloc(8 * 256 * 4);
  float* a3     = (float*)alloc(8 * 256 * 4);

  _Float16* bufA  = (_Float16*)alloc((size_t)ME * 512 * 2);  // edge inputs / 2nd activ
  float*    bufH  = (float*)   alloc((size_t)ME * 256 * 4);  // raw GEMM out (also lin1 H)
  _Float16* bufA2 = (_Float16*)alloc((size_t)ME * 256 * 2);  // 1st activ / lin1 h

  // --- KNN + conversions -------------------------------------------------
  knn_kernel<<<dim3(NP / 256, BC), 256, 0, stream>>>(pos, idx);
  cvt_f32_to_f16<<<cdiv(MP * 3, 256), 256, 0, stream>>>(pos, x0, MP * 3);
  for (int l = 0; l < 4; ++l) {
    cvt_weight_t<<<cdiv((long long)cout[l] * kp1[l], 256), 256, 0, stream>>>(
        (const float*)d_in[2 + 8 * l + 0], w1t[l], 2 * cin[l], cout[l], kp1[l]);
    cvt_weight_t<<<cdiv((long long)cout[l] * cout[l], 256), 256, 0, stream>>>(
        (const float*)d_in[2 + 8 * l + 4], w2t[l], cout[l], cout[l], cout[l]);
  }
  cvt_weight_t<<<cdiv(1024LL * 512, 256), 256, 0, stream>>>(
      (const float*)d_in[34], wl1t, 512, 1024, 512);

  auto gemm = [&](const _Float16* Ain, const _Float16* Wtp, const float* bp,
                  float* Co, long long M_, int Nd_, int Kd_) {
    dim3 grid((unsigned)(Nd_ / 64), (unsigned)(M_ / 128));
    gemm_f16_wmma<<<grid, 256, 0, stream>>>(Ain, Wtp, bp, Co, (int)M_, Nd_, Kd_);
  };
  auto bn_seq = [&](const float* H, const float* g, const float* be,
                    _Float16* out, long long M_, int C_) {
    bn_stat_partial<<<dim3(cdiv(C_, 256), NSPLIT), 256, 0, stream>>>(H, psum, pss, (int)M_, C_);
    bn_finalize<<<cdiv(C_, 256), 256, 0, stream>>>(psum, pss, meanb, istdb, (int)M_, C_);
    const long long total = M_ * C_;
    bn_apply_lrelu_f16<<<cdiv(total, 256), 256, 0, stream>>>(H, meanb, istdb, g, be, out, total, C_);
  };

  // --- Edge-conv blocks ---------------------------------------------------
  int xcOff = 0;
  const _Float16* xin = x0;
  for (int l = 0; l < 4; ++l) {
    const int Cpad = kp1[l];
    const float* b1  = (const float*)d_in[2 + 8 * l + 1];
    const float* g1  = (const float*)d_in[2 + 8 * l + 2];
    const float* be1 = (const float*)d_in[2 + 8 * l + 3];
    const float* b2  = (const float*)d_in[2 + 8 * l + 5];
    const float* g2  = (const float*)d_in[2 + 8 * l + 6];
    const float* be2 = (const float*)d_in[2 + 8 * l + 7];

    build_edges<<<cdiv(ME * Cpad, 256), 256, 0, stream>>>(xin, idx, bufA, cin[l], Cpad);
    gemm(bufA, w1t[l], b1, bufH, ME, cout[l], Cpad);
    bn_seq(bufH, g1, be1, bufA2, ME, cout[l]);
    gemm(bufA2, w2t[l], b2, bufH, ME, cout[l], cout[l]);
    bn_seq(bufH, g2, be2, bufA, ME, cout[l]);   // reuse bufA for edge activations
    max_over_k<<<cdiv(MP * (long long)cout[l], 256), 256, 0, stream>>>(
        bufA, xbuf, xc, cout[l], xcOff);
    xcOff += cout[l];
    xin = xbuf;
  }

  // --- lin1 + BN + pool ---------------------------------------------------
  gemm(xc, wl1t, (const float*)d_in[35], bufH, MP, 1024, 512);
  bn_seq(bufH, (const float*)d_in[36], (const float*)d_in[37], bufA2, MP, 1024);
  pool_max_mean<<<dim3(1024 / 256, BC), 256, 0, stream>>>(bufA2, pooled);

  // --- tail MLP (M = 8) ---------------------------------------------------
  lin_small<<<cdiv(8 * 512, 256), 256, 0, stream>>>(
      pooled, (const float*)d_in[38], (const float*)d_in[39], z2, 2048, 512);
  bn_tail_apply<<<cdiv(512, 256), 256, 0, stream>>>(
      z2, (const float*)d_in[40], (const float*)d_in[41], a2, 512);
  lin_small<<<cdiv(8 * 256, 256), 256, 0, stream>>>(
      a2, (const float*)d_in[42], (const float*)d_in[43], z3, 512, 256);
  bn_tail_apply<<<1, 256, 0, stream>>>(
      z3, (const float*)d_in[44], (const float*)d_in[45], a3, 256);
  lin_small<<<cdiv(8 * 40, 256), 256, 0, stream>>>(
      a3, (const float*)d_in[46], (const float*)d_in[47], (float*)d_out, 256, 40);
}
